// GVPConv_80307298500617
// MI455X (gfx1250) — compile-verified, matrix operand order checked by reading, not run on hardware
//
#include <hip/hip_runtime.h>
#include <hip/hip_bf16.h>
#include <math.h>

typedef __bf16 bf16;
typedef __attribute__((ext_vector_type(16))) __bf16 v16bf;
typedef __attribute__((ext_vector_type(8)))  float  v8f;

#define NN 50000
#define EE 512000
#define SS 128     // scalar channels
#define VV 16      // vector channels
#define HM 17      // message hidden vec dim (V+1)
#define KA_M 192   // padded K for message GEMM (161 -> 192)
#define KA_U 160   // padded K for update GEMM  (144 -> 160)

// ---------------------------------------------------------------------------
// WMMA fragment loader.
// 16-bit A-matrix 16x32 layout (ISA 7.12.2): lane = row (mod 16), lane-half
// selects K offsets {0..7,16..23} (half=0) vs {8..15,24..31} (half=1).
// B (K x N) uses the mirrored layout: lane = column, same K element mapping.
// `row` must point at a row-major row of stride >= kbase+32.
// ---------------------------------------------------------------------------
__device__ __forceinline__ v16bf frag_row(const bf16* row, int kbase, int half) {
  const bf16* p = row + kbase + (half << 3);
  v16bf a;
#pragma unroll
  for (int i = 0; i < 8; ++i) { a[i] = p[i]; a[i + 8] = p[i + 16]; }
  return a;
}

// v_rcp_f32 (~1 ulp) instead of the exact-division macro sequence
__device__ __forceinline__ float fast_rcp(float x) { return __builtin_amdgcn_rcpf(x); }
__device__ __forceinline__ float silu_f(float x) { return x * fast_rcp(1.0f + __expf(-x)); }
__device__ __forceinline__ float sigm_f(float x) { return fast_rcp(1.0f + __expf(-x)); }

// ---------------------------------------------------------------------------
// Prologue: zero aggregation buffers
// ---------------------------------------------------------------------------
__global__ void zero_f32_kernel(float* __restrict__ p, long n) {
  for (long i = (long)blockIdx.x * blockDim.x + threadIdx.x; i < n;
       i += (long)gridDim.x * blockDim.x)
    p[i] = 0.0f;
}

// Prologue: pack weights to bf16 (with K padding)
__global__ void pack_weights_kernel(const float* __restrict__ Wout_m,
                                    const float* __restrict__ Wg_m,
                                    const float* __restrict__ Wout_u,
                                    const float* __restrict__ Wg_u,
                                    bf16* __restrict__ Wm, bf16* __restrict__ Wg,
                                    bf16* __restrict__ Wu, bf16* __restrict__ Wgu) {
  const int n0 = SS * KA_M, n1 = VV * SS, n2 = SS * KA_U, n3 = VV * SS;
  int idx = blockIdx.x * blockDim.x + threadIdx.x;
  if (idx < n0) {
    int r = idx / KA_M, k = idx % KA_M;
    Wm[idx] = (bf16)(k < 161 ? Wout_m[r * 161 + k] : 0.0f);
  } else if (idx < n0 + n1) {
    int q = idx - n0;
    Wg[q] = (bf16)Wg_m[q];
  } else if (idx < n0 + n1 + n2) {
    int q = idx - n0 - n1;
    int r = q / KA_U, k = q % KA_U;
    Wu[q] = (bf16)(k < 144 ? Wout_u[r * 144 + k] : 0.0f);
  } else if (idx < n0 + n1 + n2 + n3) {
    int q = idx - n0 - n1 - n2;
    Wgu[q] = (bf16)Wg_u[q];
  }
}

// ---------------------------------------------------------------------------
// Edge message kernel: 16 edges per 256-thread block (8 wave32).
// ---------------------------------------------------------------------------
__global__ void __launch_bounds__(256) edge_msg_kernel(
    const float* __restrict__ ns, const float* __restrict__ nv,
    const float* __restrict__ pos, const int* __restrict__ eidx,
    const float* __restrict__ Wh_m, const float* __restrict__ Wu_m,
    const bf16* __restrict__ WmB, const float* __restrict__ bout_m,
    const bf16* __restrict__ WgB, const float* __restrict__ bg_m,
    float* __restrict__ agg_s, float* __restrict__ agg_v) {
  __shared__ bf16  Asub[16][KA_M];       // bf16 A-tile: [scal 0..127 | rbf 128..143 | sh 144..160 | pad]
  __shared__ float mvS[16][HM][3];       // message vectors (x_unit ; node_vectors[src])
  __shared__ float VhE[16][HM][3];
  __shared__ float VuE[16][VV][3];
  __shared__ bf16  featsB[16][SS];       // post-SiLU feats, bf16, for gate WMMA
  __shared__ float geoD[16];             // dij
  __shared__ int   dstS[16];

  const int t = threadIdx.x;
  const int slot = t >> 4;               // edge slot 0..15
  const int j = t & 15;                  // helper lane 0..15
  const int lane = t & 31;
  const int wave = t >> 5;
  const int half = (lane >> 4) & 1;
  const int col16 = lane & 15;

  const long e = (long)blockIdx.x * 16 + slot;
  const bool valid = (e < EE);
  const int si = valid ? eidx[e] : 0;
  const int di = valid ? eidx[EE + e] : -1;

  // ---- Phase A: gather + geometry + small vector mixes ----
  if (j == 0) {
    dstS[slot] = di;
    float dx = 0.f, dy = 0.f, dz = 0.f, dij = 1.0f;
    if (valid) {
      dx = pos[(size_t)di * 3 + 0] - pos[(size_t)si * 3 + 0];
      dy = pos[(size_t)di * 3 + 1] - pos[(size_t)si * 3 + 1];
      dz = pos[(size_t)di * 3 + 2] - pos[(size_t)si * 3 + 2];
      float d2 = fmaxf(dx * dx + dy * dy + dz * dz, 1e-8f);
      dij = sqrtf(d2);
    }
    geoD[slot] = dij;
    float rij = fast_rcp(dij);
    mvS[slot][0][0] = dx * rij;
    mvS[slot][0][1] = dy * rij;
    mvS[slot][0][2] = dz * rij;
  }
#pragma unroll
  for (int k = 0; k < 8; ++k) {
    int c = j + 16 * k;
    float v = valid ? ns[(size_t)si * SS + c] : 0.0f;
    Asub[slot][c] = (bf16)v;
  }
#pragma unroll
  for (int c = 0; c < 3; ++c)
    mvS[slot][1 + j][c] = valid ? nv[((size_t)si * VV + j) * 3 + c] : 0.0f;
  // zero K padding (cols 161..191)
  Asub[slot][161 + j] = (bf16)0.0f;
  if (177 + j < KA_M) Asub[slot][177 + j] = (bf16)0.0f;
  __syncthreads();

  // ---- RBF (cols 128..143) + Vh (cols 144..160 = ||Vh||) ----
  {
    float dij = geoD[slot];
    float mu = (20.0f / 15.0f) * (float)j;
    float td = (dij - mu) * 0.8f;        // 1/sigma, sigma = 20/16
    Asub[slot][SS + j] = (bf16)__expf(-td * td);
  }
  {
    float a0 = 0.f, a1 = 0.f, a2 = 0.f;
#pragma unroll
    for (int v = 0; v < HM; ++v) {
      float w = Wh_m[v * HM + j];
      a0 += mvS[slot][v][0] * w;
      a1 += mvS[slot][v][1] * w;
      a2 += mvS[slot][v][2] * w;
    }
    VhE[slot][j][0] = a0; VhE[slot][j][1] = a1; VhE[slot][j][2] = a2;
    Asub[slot][144 + j] = (bf16)sqrtf(fmaxf(a0 * a0 + a1 * a1 + a2 * a2, 1e-8f));
    if (j == 0) {  // h = 16
      float b0 = 0.f, b1 = 0.f, b2 = 0.f;
#pragma unroll
      for (int v = 0; v < HM; ++v) {
        float w = Wh_m[v * HM + 16];
        b0 += mvS[slot][v][0] * w;
        b1 += mvS[slot][v][1] * w;
        b2 += mvS[slot][v][2] * w;
      }
      VhE[slot][16][0] = b0; VhE[slot][16][1] = b1; VhE[slot][16][2] = b2;
      Asub[slot][160] = (bf16)sqrtf(fmaxf(b0 * b0 + b1 * b1 + b2 * b2, 1e-8f));
    }
  }
  __syncthreads();

  // ---- Vu (u = j) ----
  {
    float a0 = 0.f, a1 = 0.f, a2 = 0.f;
#pragma unroll
    for (int h = 0; h < HM; ++h) {
      float w = Wu_m[h * VV + j];
      a0 += VhE[slot][h][0] * w;
      a1 += VhE[slot][h][1] * w;
      a2 += VhE[slot][h][2] * w;
    }
    VuE[slot][j][0] = a0; VuE[slot][j][1] = a1; VuE[slot][j][2] = a2;
  }
  __syncthreads();

  // ---- Phase B: main GEMM [16 x 192] x [192 x 128] via WMMA, per-wave tile ----
  {
    v8f acc = {};
#pragma unroll
    for (int kk = 0; kk < KA_M / 32; ++kk) {
      int kb = kk * 32;
      v16bf a = frag_row(&Asub[col16][0], kb, half);
      v16bf b = frag_row(WmB + (size_t)(wave * 16 + col16) * KA_M, kb, half);
      acc = __builtin_amdgcn_wmma_f32_16x16x32_bf16(false, a, false, b,
                                                    (short)0, acc, false, false);
    }
    int col = wave * 16 + col16;
    float bias = bout_m[col];
#pragma unroll
    for (int r = 0; r < 8; ++r) {
      int es = r + (half << 3);
      float f = silu_f(acc[r] + bias);
      featsB[es][col] = (bf16)f;
      int d = dstS[es];
      if (d >= 0) atomicAdd(&agg_s[(size_t)d * SS + col], f * 0.1f);
    }
  }
  __syncthreads();

  // ---- Phase C (wave 0): gate GEMM [16 x 128] x [128 x 16] + vector scatter ----
  if (wave == 0) {
    v8f g = {};
#pragma unroll
    for (int kk = 0; kk < SS / 32; ++kk) {
      int kb = kk * 32;
      v16bf a = frag_row(&featsB[col16][0], kb, half);
      v16bf b = frag_row(WgB + (size_t)col16 * SS, kb, half);
      g = __builtin_amdgcn_wmma_f32_16x16x32_bf16(false, a, false, b,
                                                  (short)0, g, false, false);
    }
    int u = col16;
    float bgv = bg_m[u];
#pragma unroll
    for (int r = 0; r < 8; ++r) {
      int es = r + (half << 3);
      float sg = sigm_f(g[r] + bgv);
      int d = dstS[es];
      if (d >= 0) {
#pragma unroll
        for (int c = 0; c < 3; ++c)
          atomicAdd(&agg_v[((size_t)d * VV + u) * 3 + c],
                    sg * VuE[es][u][c] * 0.1f);
      }
    }
  }
}

// ---------------------------------------------------------------------------
// Node update kernel: 16 nodes per 256-thread block.
// residual + LN1 -> update GVP (WMMA) -> residual + LN2 -> outputs
// ---------------------------------------------------------------------------
__global__ void __launch_bounds__(256) node_update_kernel(
    const float* __restrict__ ns, const float* __restrict__ nv,
    const float* __restrict__ agg_s, const float* __restrict__ agg_v,
    const float* __restrict__ Wh_u, const float* __restrict__ Wu_u,
    const bf16* __restrict__ WuB, const float* __restrict__ bout_u,
    const bf16* __restrict__ WguB, const float* __restrict__ bg_u,
    const float* __restrict__ g1, const float* __restrict__ b1,
    const float* __restrict__ g2, const float* __restrict__ b2,
    float* __restrict__ outS, float* __restrict__ outV) {
  __shared__ bf16  Asub[16][KA_U];
  __shared__ float sF[16][SS];           // s (then s after LN1)
  __shared__ float vV[16][VV][3];        // v (then v/vn, then v2)
  __shared__ float VhU[16][VV][3];
  __shared__ float VuU[16][VV][3];       // Vu (then sigmoid-gated uv)
  __shared__ bf16  featsB[16][SS];
  __shared__ float usF[16][SS];
  __shared__ float red[16][16], red2[16][16], qred[16][16];
  __shared__ float stats[16][3];

  const int t = threadIdx.x;
  const int slot = t >> 4;
  const int j = t & 15;
  const int lane = t & 31;
  const int wave = t >> 5;
  const int half = (lane >> 4) & 1;
  const int col16 = lane & 15;

  const long i = (long)blockIdx.x * 16 + slot;
  const bool valid = (i < NN);

  // ---- residual + partial LN1 stats ----
  {
    float ps = 0.f, ps2 = 0.f;
#pragma unroll
    for (int k = 0; k < 8; ++k) {
      int c = j + 16 * k;
      float x = valid ? (ns[(size_t)i * SS + c] + agg_s[(size_t)i * SS + c]) : 0.0f;
      sF[slot][c] = x;
      ps += x; ps2 += x * x;
    }
    float q = 0.f;
#pragma unroll
    for (int c = 0; c < 3; ++c) {
      float x = valid ? (nv[((size_t)i * VV + j) * 3 + c] +
                         agg_v[((size_t)i * VV + j) * 3 + c]) : 0.0f;
      vV[slot][j][c] = x;
      q += x * x;
    }
    red[slot][j] = ps; red2[slot][j] = ps2; qred[slot][j] = fmaxf(q, 1e-8f);
  }
  __syncthreads();
  if (j == 0) {
    float s = 0.f, s2 = 0.f, qs = 0.f;
#pragma unroll
    for (int k = 0; k < 16; ++k) { s += red[slot][k]; s2 += red2[slot][k]; qs += qred[slot][k]; }
    float mean = s / (float)SS;
    float var = fmaxf(s2 / (float)SS - mean * mean, 0.0f);
    stats[slot][0] = mean;
    stats[slot][1] = rsqrtf(var + 1e-5f);
    stats[slot][2] = fast_rcp(sqrtf(qs / (float)VV + 1e-5f) + 1e-5f);  // 1/vn
  }
  __syncthreads();

  // ---- apply LN1; build A-tile scalars; zero pad ----
  {
    float mean = stats[slot][0], rstd = stats[slot][1], vninv = stats[slot][2];
#pragma unroll
    for (int k = 0; k < 8; ++k) {
      int c = j + 16 * k;
      float x = (sF[slot][c] - mean) * rstd * g1[c] + b1[c];
      sF[slot][c] = x;
      Asub[slot][c] = (bf16)x;
    }
#pragma unroll
    for (int c = 0; c < 3; ++c) vV[slot][j][c] = vV[slot][j][c] * vninv;
    Asub[slot][144 + j] = (bf16)0.0f;    // pad 144..159
  }
  __syncthreads();

  // ---- Vh (h = j), sh into A cols 128..143 ----
  {
    float a0 = 0.f, a1 = 0.f, a2 = 0.f;
#pragma unroll
    for (int v = 0; v < VV; ++v) {
      float w = Wh_u[v * VV + j];
      a0 += vV[slot][v][0] * w;
      a1 += vV[slot][v][1] * w;
      a2 += vV[slot][v][2] * w;
    }
    VhU[slot][j][0] = a0; VhU[slot][j][1] = a1; VhU[slot][j][2] = a2;
    Asub[slot][SS + j] = (bf16)sqrtf(fmaxf(a0 * a0 + a1 * a1 + a2 * a2, 1e-8f));
  }
  __syncthreads();

  // ---- Vu (u = j) ----
  {
    float a0 = 0.f, a1 = 0.f, a2 = 0.f;
#pragma unroll
    for (int h = 0; h < VV; ++h) {
      float w = Wu_u[h * VV + j];
      a0 += VhU[slot][h][0] * w;
      a1 += VhU[slot][h][1] * w;
      a2 += VhU[slot][h][2] * w;
    }
    VuU[slot][j][0] = a0; VuU[slot][j][1] = a1; VuU[slot][j][2] = a2;
  }
  __syncthreads();

  // ---- main GEMM [16 x 160] x [160 x 128] via WMMA ----
  {
    v8f acc = {};
#pragma unroll
    for (int kk = 0; kk < KA_U / 32; ++kk) {
      int kb = kk * 32;
      v16bf a = frag_row(&Asub[col16][0], kb, half);
      v16bf b = frag_row(WuB + (size_t)(wave * 16 + col16) * KA_U, kb, half);
      acc = __builtin_amdgcn_wmma_f32_16x16x32_bf16(false, a, false, b,
                                                    (short)0, acc, false, false);
    }
    int col = wave * 16 + col16;
    float bias = bout_u[col];
#pragma unroll
    for (int r = 0; r < 8; ++r) {
      int es = r + (half << 3);
      float f = silu_f(acc[r] + bias);
      usF[es][col] = f;
      featsB[es][col] = (bf16)f;
    }
  }
  __syncthreads();

  // ---- gate GEMM (wave 0), scale Vu in place ----
  if (wave == 0) {
    v8f g = {};
#pragma unroll
    for (int kk = 0; kk < SS / 32; ++kk) {
      int kb = kk * 32;
      v16bf a = frag_row(&featsB[col16][0], kb, half);
      v16bf b = frag_row(WguB + (size_t)col16 * SS, kb, half);
      g = __builtin_amdgcn_wmma_f32_16x16x32_bf16(false, a, false, b,
                                                  (short)0, g, false, false);
    }
    int u = col16;
    float bgv = bg_u[u];
#pragma unroll
    for (int r = 0; r < 8; ++r) {
      int es = r + (half << 3);
      float sg = sigm_f(g[r] + bgv);
#pragma unroll
      for (int c = 0; c < 3; ++c) VuU[es][u][c] *= sg;
    }
  }
  __syncthreads();

  // ---- residual + LN2 partials ----
  {
    float ps = 0.f, ps2 = 0.f;
#pragma unroll
    for (int k = 0; k < 8; ++k) {
      int c = j + 16 * k;
      float x = sF[slot][c] + usF[slot][c];
      ps += x; ps2 += x * x;
    }
    float q = 0.f;
#pragma unroll
    for (int c = 0; c < 3; ++c) {
      float v2 = vV[slot][j][c] + VuU[slot][j][c];
      vV[slot][j][c] = v2;
      q += v2 * v2;
    }
    red[slot][j] = ps; red2[slot][j] = ps2; qred[slot][j] = fmaxf(q, 1e-8f);
  }
  __syncthreads();
  if (j == 0) {
    float s = 0.f, s2 = 0.f, qs = 0.f;
#pragma unroll
    for (int k = 0; k < 16; ++k) { s += red[slot][k]; s2 += red2[slot][k]; qs += qred[slot][k]; }
    float mean = s / (float)SS;
    float var = fmaxf(s2 / (float)SS - mean * mean, 0.0f);
    stats[slot][0] = mean;
    stats[slot][1] = rsqrtf(var + 1e-5f);
    stats[slot][2] = fast_rcp(sqrtf(qs / (float)VV + 1e-5f) + 1e-5f);  // 1/vn
  }
  __syncthreads();

  if (valid) {
    float mean = stats[slot][0], rstd = stats[slot][1], vninv = stats[slot][2];
#pragma unroll
    for (int k = 0; k < 8; ++k) {
      int c = j + 16 * k;
      float x = sF[slot][c] + usF[slot][c];
      outS[(size_t)i * SS + c] = (x - mean) * rstd * g2[c] + b2[c];
    }
#pragma unroll
    for (int c = 0; c < 3; ++c)
      outV[((size_t)i * VV + j) * 3 + c] = vV[slot][j][c] * vninv;
  }
}

// ---------------------------------------------------------------------------
// Host-side launcher
// ---------------------------------------------------------------------------
extern "C" void kernel_launch(void* const* d_in, const int* in_sizes, int n_in,
                              void* d_out, int out_size, void* d_ws, size_t ws_size,
                              hipStream_t stream) {
  (void)in_sizes; (void)n_in; (void)out_size; (void)ws_size;

  const float* ns     = (const float*)d_in[0];
  const float* nv     = (const float*)d_in[1];
  const float* pos    = (const float*)d_in[2];
  const int*   eidx   = (const int*)  d_in[3];
  const float* Wh_m   = (const float*)d_in[4];
  const float* Wu_m   = (const float*)d_in[5];
  const float* Wout_m = (const float*)d_in[6];
  const float* bout_m = (const float*)d_in[7];
  const float* Wg_m   = (const float*)d_in[8];
  const float* bg_m   = (const float*)d_in[9];
  const float* Wh_u   = (const float*)d_in[10];
  const float* Wu_u   = (const float*)d_in[11];
  const float* Wout_u = (const float*)d_in[12];
  const float* bout_u = (const float*)d_in[13];
  const float* Wg_u   = (const float*)d_in[14];
  const float* bg_u   = (const float*)d_in[15];
  const float* ln1_g  = (const float*)d_in[16];
  const float* ln1_b  = (const float*)d_in[17];
  const float* ln2_g  = (const float*)d_in[18];
  const float* ln2_b  = (const float*)d_in[19];

  // workspace layout
  float* agg_s = (float*)d_ws;                      // N*128 f32
  float* agg_v = agg_s + (size_t)NN * SS;           // N*48  f32
  bf16*  WmB   = (bf16*)(agg_v + (size_t)NN * VV * 3);
  bf16*  WgB   = WmB + (size_t)SS * KA_M;
  bf16*  WuB   = WgB + (size_t)VV * SS;
  bf16*  WguB  = WuB + (size_t)SS * KA_U;

  // 1) zero aggregation buffers (must happen every call)
  zero_f32_kernel<<<4096, 256, 0, stream>>>(agg_s, (long)NN * (SS + VV * 3));

  // 2) pack weights to bf16
  const int packTotal = SS * KA_M + VV * SS + SS * KA_U + VV * SS;
  pack_weights_kernel<<<(packTotal + 255) / 256, 256, 0, stream>>>(
      Wout_m, Wg_m, Wout_u, Wg_u, WmB, WgB, WuB, WguB);

  // 3) per-edge messages + scatter
  edge_msg_kernel<<<(EE + 15) / 16, 256, 0, stream>>>(
      ns, nv, pos, eidx, Wh_m, Wu_m, WmB, bout_m, WgB, bg_m, agg_s, agg_v);

  // 4) node update + layernorms -> outputs
  float* outS = (float*)d_out;
  float* outV = outS + (size_t)NN * SS;
  node_update_kernel<<<(NN + 15) / 16, 256, 0, stream>>>(
      ns, nv, agg_s, agg_v, Wh_u, Wu_u, WuB, bout_u, WguB, bg_u,
      ln1_g, ln1_b, ln2_g, ln2_b, outS, outV);
}
